// PraxisPEER_53068615909868
// MI455X (gfx1250) — matrix operand order repaired and checked by reading, not compiled, not run on hardware
//
#include <hip/hip_runtime.h>
#include <hip/hip_bf16.h>
#include <math.h>

// ---------------------------------------------------------------------------
// PEER / product-key mixture-of-experts forward for MI455X (gfx1250, wave32).
// Fixed problem: B=2,N=1024 -> T=2048 tokens, D=1024, H=8, KD=128,
// NUM_KEYS=256, K=16, E=65536.
// GEMMs run as bf16 WMMA (v_wmma_f32_16x16x32_bf16, fp32 accum).
// Expert gather (~2.1 GB) is the roofline bottleneck -> coalesced float4,
// with block-uniform staging done via GLOBAL_LOAD_ASYNC_TO_LDS (ASYNCcnt).
// ---------------------------------------------------------------------------

#define T_TOK   2048
#define DIM     1024
#define NHEAD   8
#define KD      128
#define NKEYS   256
#define TOPK    16
#define QCOLS   2048   // 2*H*KD

typedef __attribute__((ext_vector_type(8)))  __bf16 v8bf;
typedef __attribute__((ext_vector_type(16))) __bf16 v16bf;
typedef __attribute__((ext_vector_type(8)))  float  v8f;
typedef __attribute__((ext_vector_type(4)))  int    v4i;

#if defined(__gfx1250__) && \
    __has_builtin(__builtin_amdgcn_global_load_async_to_lds_b128) && \
    __has_builtin(__builtin_amdgcn_s_wait_asynccnt)
#define USE_ASYNC_LDS 1
#else
#define USE_ASYNC_LDS 0
#endif

#if USE_ASYNC_LDS
// Builtin signature (from the round-2 diagnostic): arg0 is
// 'int __vector(4) __device__ *'  ==  v4i in address_space(1).
// arg1 is the LDS destination     ==  v4i in address_space(3).
#define GPTR(p) ((__attribute__((address_space(1))) v4i*)(p))
#define LPTR(p) ((__attribute__((address_space(3))) v4i*)(p))
#endif

// ---------------------------------------------------------------------------
// 1) BatchNorm statistics: per-dim mean / rstd over 2048 tokens.
// ---------------------------------------------------------------------------
__global__ void bn_stats_kernel(const float* __restrict__ x,
                                float* __restrict__ mean,
                                float* __restrict__ rstd) {
    int d = blockIdx.x * 256 + threadIdx.x;            // 0..1023
    float s = 0.f, s2 = 0.f;
    for (int t = 0; t < T_TOK; ++t) {
        float v = x[(size_t)t * DIM + d];              // coalesced across d
        s += v; s2 += v * v;
    }
    float m   = s  * (1.0f / T_TOK);
    float var = s2 * (1.0f / T_TOK) - m * m;
    mean[d] = m;
    rstd[d] = rsqrtf(var + 1e-5f);
}

// ---------------------------------------------------------------------------
// 2) BN apply + cast to bf16:  xn[t][d]
// ---------------------------------------------------------------------------
__global__ void bn_apply_kernel(const float* __restrict__ x,
                                const float* __restrict__ mean,
                                const float* __restrict__ rstd,
                                const float* __restrict__ gamma,
                                const float* __restrict__ beta,
                                __bf16* __restrict__ xn) {
    size_t i = (size_t)blockIdx.x * 256 + threadIdx.x; // 2M elements
    int d = (int)(i & (DIM - 1));
    float v = (x[i] - mean[d]) * rstd[d] * gamma[d] + beta[d];
    xn[i] = (__bf16)v;
}

// ---------------------------------------------------------------------------
// 3) Transpose + cast Wq [1024][2048] f32 -> Wqt [2048][1024] bf16
//    (K-contiguous so WMMA B fragments load 16B chunks).
// ---------------------------------------------------------------------------
__global__ void cast_wq_kernel(const float* __restrict__ Wq,
                               __bf16* __restrict__ Wqt) {
    size_t i = (size_t)blockIdx.x * 256 + threadIdx.x; // 2M elements, i=k*2048+n
    int n = (int)(i & (QCOLS - 1));
    int k = (int)(i >> 11);
    Wqt[(size_t)n * DIM + k] = (__bf16)Wq[i];
}

// ---------------------------------------------------------------------------
// 4) Reorder + cast keys [H][256][2][128] f32 -> [p*8+h][256][128] bf16
// ---------------------------------------------------------------------------
__global__ void cast_keys_kernel(const float* __restrict__ keys,
                                 __bf16* __restrict__ kb) {
    size_t i = (size_t)blockIdx.x * 256 + threadIdx.x; // 524288 outputs
    int kd  = (int)(i & (KD - 1));
    int key = (int)((i >> 7) & (NKEYS - 1));
    int ph  = (int)(i >> 15);
    int p = ph >> 3, h = ph & 7;
    kb[i] = (__bf16)keys[(((size_t)h * NKEYS + key) * 2 + p) * KD + kd];
}

// ---------------------------------------------------------------------------
// WMMA fragment helpers (ISA 7.12.2 layouts, wave32).
//   A 16x32 bf16: lane<16 -> M=lane,   K in {k0..k0+7, k0+16..k0+23}
//                 lane>=16 -> M=lane-16, K in {k0+8..k0+15, k0+24..k0+31}
//   B 32x16 bf16: N=lane&15; K order same as A (lane<16 -> k0..k0+7, k0+16..k0+23; lane>=16 -> k0+8..k0+15, k0+24..k0+31)
// Both operands come from K-contiguous rows -> two 16B loads each.
// ---------------------------------------------------------------------------
__device__ __forceinline__ v16bf load_a_frag(const __bf16* rowp, int k0, int hi) {
    v8bf lo = *(const v8bf*)(rowp + k0 + hi * 8);
    v8bf hi8 = *(const v8bf*)(rowp + k0 + hi * 8 + 16);
    v16bf a;
#pragma unroll
    for (int i = 0; i < 8; ++i) { a[i] = lo[i]; a[i + 8] = hi8[i]; }
    return a;
}
__device__ __forceinline__ v16bf load_b_frag(const __bf16* colp, int k0, int hi) {
    v8bf lo = *(const v8bf*)(colp + k0 + hi * 8);
    v8bf hi8 = *(const v8bf*)(colp + k0 + hi * 8 + 16);
    v16bf b;
#pragma unroll
    for (int i = 0; i < 8; ++i) { b[i] = lo[i]; b[i + 8] = hi8[i]; }
    return b;
}

// ---------------------------------------------------------------------------
// 5) q = xn @ Wq  (M=2048, N=2048, K=1024), bf16 WMMA, fp32 accum -> bf16 out.
//    Block = 128 threads = 4 waves; wave computes one 16x16 tile.
// ---------------------------------------------------------------------------
__global__ void qgemm_kernel(const __bf16* __restrict__ A,   // [2048][1024]
                             const __bf16* __restrict__ Bt,  // [2048][1024]
                             __bf16* __restrict__ Q) {       // [2048][2048]
    const int lane = threadIdx.x & 31;
    const int wv   = threadIdx.x >> 5;
    const int n0   = (blockIdx.x * 4 + wv) * 16;
    const int m0   = blockIdx.y * 16;
    const int hi   = lane >> 4;
    const int l15  = lane & 15;
    const __bf16* arow = A  + (size_t)(m0 + l15) * DIM;
    const __bf16* bcol = Bt + (size_t)(n0 + l15) * DIM;
    v8f c = {};
#pragma unroll 4
    for (int k0 = 0; k0 < DIM; k0 += 32) {
        v16bf a = load_a_frag(arow, k0, hi);
        v16bf b = load_b_frag(bcol, k0, hi);
        c = __builtin_amdgcn_wmma_f32_16x16x32_bf16(
                false, a, false, b, (short)0, c, false, false);
    }
    const int r0 = m0 + hi * 8;
#pragma unroll
    for (int i = 0; i < 8; ++i)
        Q[(size_t)(r0 + i) * QCOLS + (n0 + l15)] = (__bf16)c[i];
}

// ---------------------------------------------------------------------------
// 6) sim[ph][token][key] = q_ph @ keys_phᵀ  (M=2048, N=256, K=128) per ph.
// ---------------------------------------------------------------------------
__global__ void simgemm_kernel(const __bf16* __restrict__ Q,    // [2048][2048]
                               const __bf16* __restrict__ Keys, // [16][256][128]
                               float* __restrict__ Sim) {       // [16][2048][256]
    const int ph   = blockIdx.z;
    const int lane = threadIdx.x & 31;
    const int wv   = threadIdx.x >> 5;
    const int n0   = (blockIdx.x * 4 + wv) * 16;
    const int m0   = blockIdx.y * 16;
    const int hi   = lane >> 4;
    const int l15  = lane & 15;
    const __bf16* arow = Q    + (size_t)(m0 + l15) * QCOLS + ph * KD;
    const __bf16* bcol = Keys + ((size_t)ph * NKEYS + (n0 + l15)) * KD;
    v8f c = {};
#pragma unroll
    for (int k0 = 0; k0 < KD; k0 += 32) {
        v16bf a = load_a_frag(arow, k0, hi);
        v16bf b = load_b_frag(bcol, k0, hi);
        c = __builtin_amdgcn_wmma_f32_16x16x32_bf16(
                false, a, false, b, (short)0, c, false, false);
    }
    const int r0 = m0 + hi * 8;
#pragma unroll
    for (int i = 0; i < 8; ++i)
        Sim[((size_t)ph * T_TOK + (r0 + i)) * NKEYS + (n0 + l15)] = c[i];
}

// ---------------------------------------------------------------------------
// Wave-wide argmax (wave32) on (value, index) pairs.
// ---------------------------------------------------------------------------
__device__ __forceinline__ void wave_argmax(float& v, int& i) {
#pragma unroll
    for (int off = 16; off > 0; off >>= 1) {
        float ov = __shfl_xor(v, off, 32);
        int   oi = __shfl_xor(i, off, 32);
        if (ov > v || (ov == v && oi < i)) { v = ov; i = oi; }
    }
}

// ---------------------------------------------------------------------------
// 7) Per-(p,h,token) top-16 of 256: one wave per problem, iterative select.
// ---------------------------------------------------------------------------
__global__ void topk1_kernel(const float* __restrict__ Sim,
                             float* __restrict__ Sx, int* __restrict__ Ix) {
    const int gw   = blockIdx.x * 8 + (threadIdx.x >> 5); // 0..32767
    const int lane = threadIdx.x & 31;
    const float4* s4 = (const float4*)(Sim + (size_t)gw * NKEYS);
    float4 v0 = s4[lane * 2];
    float4 v1 = s4[lane * 2 + 1];
    float vals[8] = { v0.x, v0.y, v0.z, v0.w, v1.x, v1.y, v1.z, v1.w };
    float* ox = Sx + (size_t)gw * TOPK;
    int*   oi = Ix + (size_t)gw * TOPK;
#pragma unroll 1
    for (int r = 0; r < TOPK; ++r) {
        float best = -3.4e38f; int bi = 0;
#pragma unroll
        for (int j = 0; j < 8; ++j)
            if (vals[j] > best) { best = vals[j]; bi = lane * 8 + j; }
        wave_argmax(best, bi);
        if ((bi >> 3) == lane) vals[bi & 7] = -3.4e38f;
        if (lane == 0) { ox[r] = best; oi[r] = bi; }
    }
}

// ---------------------------------------------------------------------------
// 8) Cartesian merge of the two halves + top-16 + sigmoid gate.
//    One wave per (token,h); lane owns 8 of 256 combos:
//    lane -> i = lane>>1 (x-slot), jy in [(lane&1)*8, +8) (y-slot).
// ---------------------------------------------------------------------------
__global__ void merge_topk_kernel(const float* __restrict__ Sx,
                                  const int* __restrict__ Ix,
                                  float* __restrict__ Gate,
                                  int* __restrict__ Eidx) {
    const int gw    = blockIdx.x * 8 + (threadIdx.x >> 5); // 0..16383
    const int lane  = threadIdx.x & 31;
    const int token = gw >> 3, h = gw & 7;
    const size_t bx = (((size_t)h)       * T_TOK + token) * TOPK; // p=0
    const size_t by = (((size_t)(8 + h)) * T_TOK + token) * TOPK; // p=1
    const int i   = lane >> 1;
    const int jy0 = (lane & 1) * 8;
    const float sxi = Sx[bx + i];
    const int   ixi = Ix[bx + i];
    float syv[8]; int iyv[8];
#pragma unroll
    for (int j = 0; j < 8; ++j) { syv[j] = Sx[by + jy0 + j]; iyv[j] = Ix[by + jy0 + j]; }
    float vals[8];
#pragma unroll
    for (int j = 0; j < 8; ++j) vals[j] = sxi + syv[j];
    float* og = Gate + ((size_t)token * NHEAD + h) * TOPK;
    int*   oe = Eidx + ((size_t)token * NHEAD + h) * TOPK;
#pragma unroll 1
    for (int r = 0; r < TOPK; ++r) {
        float best = -3.4e38f; int bc = 0;
#pragma unroll
        for (int j = 0; j < 8; ++j)
            if (vals[j] > best) { best = vals[j]; bc = lane * 8 + j; }
        wave_argmax(best, bc);
        if ((bc >> 3) == lane) {
            int j = bc & 7;
            vals[j] = -3.4e38f;
            og[r] = 1.0f / (1.0f + __expf(-best));
            oe[r] = ixi * NKEYS + iyv[j];
        }
    }
}

// ---------------------------------------------------------------------------
// 9) Expert gather + apply (memory-roofline kernel, ~2.1 GB of gathers).
//    Block-uniform staging (x row, indices, gates) goes through the
//    GLOBAL_LOAD_ASYNC_TO_LDS path (ASYNCcnt) when available; the per-lane
//    gather loads stay in VGPRs (no cross-lane reuse inside a row pass).
//    Phase 1: 8 waves x 16 experts compute h_k = <x, down_w[idx_k]> with
//    float4 loads + wave reduce, then g_k = sigmoid(score)*gelu(h_k).
//    Phase 2: 256 threads each own one float4 of the output row and
//    accumulate over all 128 experts from up_w (fully coalesced 16B loads).
// ---------------------------------------------------------------------------
__global__ void expert_kernel(const float* __restrict__ x,
                              const float* __restrict__ Gate,
                              const int*   __restrict__ Eidx,
                              const float* __restrict__ down_w,
                              const float* __restrict__ up_w,
                              float* __restrict__ out) {
    __shared__ float4 s_x[DIM / 4];
    __shared__ float  s_g[NHEAD * TOPK];
    __shared__ float  s_gt[NHEAD * TOPK];
    __shared__ int    s_idx[NHEAD * TOPK];
    const int token = blockIdx.x;
    const int t = threadIdx.x;
    const float* xrow = x + (size_t)token * DIM;

#if USE_ASYNC_LDS
    // Async staging: 256 threads x 16B x-row, 32 threads x 16B indices,
    // 32 threads x 16B gates.  (global src v4i AS1*, lds dst v4i AS3*,
    //  imm offset, cpol)
    __builtin_amdgcn_global_load_async_to_lds_b128(
        GPTR(xrow + t * 4), LPTR(&s_x[t]), 0, 0);
    if (t < 32) {
        __builtin_amdgcn_global_load_async_to_lds_b128(
            GPTR(Eidx + (size_t)token * (NHEAD * TOPK) + t * 4),
            LPTR(&s_idx[t * 4]), 0, 0);
    } else if (t < 64) {
        __builtin_amdgcn_global_load_async_to_lds_b128(
            GPTR(Gate + (size_t)token * (NHEAD * TOPK) + (t - 32) * 4),
            LPTR(&s_gt[(t - 32) * 4]), 0, 0);
    }
    __builtin_amdgcn_s_wait_asynccnt(0);
    __syncthreads();
#else
    s_x[t] = ((const float4*)xrow)[t];
    if (t < NHEAD * TOPK) {
        s_idx[t] = Eidx[(size_t)token * (NHEAD * TOPK) + t];
        s_gt[t]  = Gate[(size_t)token * (NHEAD * TOPK) + t];
    }
    __syncthreads();
#endif

    const int wave = t >> 5, lane = t & 31;
#pragma unroll 1
    for (int e = 0; e < TOPK; ++e) {
        const int k = wave * TOPK + e;                 // wave == head
        const int idx = s_idx[k];
        const float4* wr = (const float4*)(down_w + (size_t)idx * DIM);
        __builtin_prefetch(up_w + (size_t)idx * DIM, 0, 1); // warm GL2 for phase 2
        float acc = 0.f;
#pragma unroll
        for (int i = 0; i < 8; ++i) {
            float4 w4 = wr[lane + 32 * i];             // coalesced 16B per lane
            float4 x4 = s_x[lane + 32 * i];
            acc = fmaf(w4.x, x4.x, fmaf(w4.y, x4.y,
                  fmaf(w4.z, x4.z, fmaf(w4.w, x4.w, acc))));
        }
#pragma unroll
        for (int off = 16; off > 0; off >>= 1) acc += __shfl_xor(acc, off, 32);
        if (lane == 0) {
            float ge = 0.5f * acc * (1.0f + erff(acc * 0.70710678118654752f));
            s_g[k] = s_gt[k] * ge;                     // sigmoid(score)*gelu(h)
        }
    }
    __syncthreads();
    float4 a4 = make_float4(0.f, 0.f, 0.f, 0.f);
#pragma unroll 1
    for (int k = 0; k < NHEAD * TOPK; ++k) {
        const float g = s_g[k];                        // LDS broadcast
        float4 u = ((const float4*)(up_w + (size_t)s_idx[k] * DIM))[t];
        a4.x += g * u.x; a4.y += g * u.y; a4.z += g * u.z; a4.w += g * u.w;
    }
    ((float4*)(out + (size_t)token * DIM))[t] = a4;
}

// ---------------------------------------------------------------------------
// Launch: inputs order = inputs, bn_gamma, bn_beta, Wq, keys_p, down_w, up_w
// ---------------------------------------------------------------------------
extern "C" void kernel_launch(void* const* d_in, const int* in_sizes, int n_in,
                              void* d_out, int out_size, void* d_ws, size_t ws_size,
                              hipStream_t stream) {
    const float* x      = (const float*)d_in[0];
    const float* gamma  = (const float*)d_in[1];
    const float* beta   = (const float*)d_in[2];
    const float* Wq     = (const float*)d_in[3];
    const float* keys_p = (const float*)d_in[4];
    const float* down_w = (const float*)d_in[5];
    const float* up_w   = (const float*)d_in[6];
    float* out = (float*)d_out;

    char* ws = (char*)d_ws;
    size_t off = 0;
    auto take = [&](size_t bytes) -> char* {
        char* p = ws + off;
        off = (off + bytes + 255) & ~(size_t)255;
        return p;
    };
    float*  mean  = (float*)take(DIM * 4);
    float*  rstd  = (float*)take(DIM * 4);
    __bf16* xn    = (__bf16*)take((size_t)T_TOK * DIM * 2);
    __bf16* wqt   = (__bf16*)take((size_t)QCOLS * DIM * 2);
    __bf16* keysb = (__bf16*)take((size_t)16 * NKEYS * KD * 2);
    __bf16* qb    = (__bf16*)take((size_t)T_TOK * QCOLS * 2);
    float*  sim   = (float*)take((size_t)16 * T_TOK * NKEYS * 4);
    float*  sx    = (float*)take((size_t)16 * T_TOK * TOPK * 4);
    int*    ix    = (int*)take((size_t)16 * T_TOK * TOPK * 4);
    float*  gate  = (float*)take((size_t)T_TOK * NHEAD * TOPK * 4);
    int*    eidx  = (int*)take((size_t)T_TOK * NHEAD * TOPK * 4);
    (void)ws_size; (void)in_sizes; (void)n_in; (void)out_size;

    bn_stats_kernel<<<DIM / 256, 256, 0, stream>>>(x, mean, rstd);
    bn_apply_kernel<<<(T_TOK * DIM) / 256, 256, 0, stream>>>(x, mean, rstd, gamma, beta, xn);
    cast_wq_kernel<<<(DIM * QCOLS) / 256, 256, 0, stream>>>(Wq, wqt);
    cast_keys_kernel<<<(16 * NKEYS * KD) / 256, 256, 0, stream>>>(keys_p, keysb);

    qgemm_kernel<<<dim3(QCOLS / 64, T_TOK / 16), 128, 0, stream>>>(xn, wqt, qb);
    simgemm_kernel<<<dim3(NKEYS / 64, T_TOK / 16, 16), 128, 0, stream>>>(qb, keysb, sim);

    topk1_kernel<<<(16 * T_TOK) / 8, 256, 0, stream>>>(sim, sx, ix);
    merge_topk_kernel<<<(T_TOK * NHEAD) / 8, 256, 0, stream>>>(sx, ix, gate, eidx);

    expert_kernel<<<T_TOK, 256, 0, stream>>>(x, gate, eidx, down_w, up_w, out);
}
